// LocalTransformer_30451318129288
// MI455X (gfx1250) — matrix operand order repaired and checked by reading, not compiled, not run on hardware
//
#include <hip/hip_runtime.h>

#define DMODEL 1024
#define NHEAD  16
#define HD     64
#define DFF    4096
#define SEQ    1024
#define BATCH  4
#define WIN    64
#define MROWS  (SEQ * BATCH)   // 4096
#define NEG_BIG (-3.0e38f)

typedef __attribute__((ext_vector_type(16))) _Float16 v16h;
typedef __attribute__((ext_vector_type(8)))  _Float16 v8h;
typedef __attribute__((ext_vector_type(8)))  float    v8f;

static __device__ __forceinline__ v8f wmma_f16(v16h a, v16h b, v8f c) {
  // D = A(16x32 f16) * B(32x16 f16) + C(16x16 f32)
  return __builtin_amdgcn_wmma_f32_16x16x32_f16(false, a, false, b, (short)0, c,
                                                false, false);
}

// ---------------------------------------------------------------------------
// CDNA5 Tensor Data Mover support (guarded; falls back to plain loads)
// ---------------------------------------------------------------------------
#if defined(__has_builtin)
#if __has_builtin(__builtin_amdgcn_tensor_load_to_lds)
#define USE_TDM 1
#endif
#endif

template <short N>
static __device__ __forceinline__ void wait_tensorcnt() {
#if defined(__has_builtin)
#if __has_builtin(__builtin_amdgcn_s_wait_tensorcnt)
  __builtin_amdgcn_s_wait_tensorcnt(N);
#else
  if (N == 0) asm volatile("s_wait_tensorcnt 0x0" ::: "memory");
  else        asm volatile("s_wait_tensorcnt 0x2" ::: "memory");
#endif
#endif
}

#ifdef USE_TDM
typedef unsigned int v4u __attribute__((ext_vector_type(4)));
typedef int          v4i __attribute__((ext_vector_type(4)));
typedef int          v8i __attribute__((ext_vector_type(8)));

// DMA a 2D f16 tile of 64 (contiguous) x 128 (rows, stride strideElems) from
// global memory into LDS at byte address `lds`. D# per CDNA5 ISA 8.3/8.4.
static __device__ __forceinline__ void tdm_load_tile_f16(
    const _Float16* g, unsigned int lds, unsigned int strideElems,
    unsigned int rows) {
  const unsigned long long ga = (unsigned long long)g;
  v4u g0;
  g0[0] = 1u;                                   // count=1, user mode, no gather
  g0[1] = lds;                                  // lds_addr (bytes)
  g0[2] = (unsigned int)ga;                     // global_addr[31:0]
  g0[3] = (unsigned int)((ga >> 32) & 0x01FFFFFFu) | (2u << 30);  // [56:32]|type=2
  v8i g1;
  g1[0] = (int)(1u << 16);                      // data_size = 1 -> 2 bytes
  g1[1] = (int)((strideElems & 0xFFFFu) << 16);               // tensor_dim0 lo16
  g1[2] = (int)((strideElems >> 16) & 0xFFFFu)                // tensor_dim0 hi16
        | (int)((rows & 0xFFFFu) << 16);                      // tensor_dim1 lo16
  g1[3] = (int)((rows >> 16) & 0xFFFFu)                       // tensor_dim1 hi16
        | (int)(64u << 16);                                   // tile_dim0 = 64
  g1[4] = 128;                                  // tile_dim1 = 128, tile_dim2 = 0
  g1[5] = (int)strideElems;                     // tensor_dim0_stride lo32
  g1[6] = 0;                                    // stride hi16 | dim1_stride lo16
  g1[7] = 0;
  v4i z4 = {0, 0, 0, 0};
#if __clang_major__ >= 23
  v8i z8 = {0, 0, 0, 0, 0, 0, 0, 0};
  __builtin_amdgcn_tensor_load_to_lds(g0, g1, z4, z4, z8, 0);
#else
  __builtin_amdgcn_tensor_load_to_lds(g0, g1, z4, z4, 0);
#endif
}
#endif  // USE_TDM

// ---------------------------------------------------------------------------
// elementwise f32 -> f16 (weights); n is a multiple of 256
// ---------------------------------------------------------------------------
__global__ __launch_bounds__(256) void f32_to_f16_kernel(
    const float* __restrict__ src, _Float16* __restrict__ dst) {
  size_t i = (size_t)blockIdx.x * 256 + threadIdx.x;
  dst[i] = (_Float16)src[i];
}

// x [S,B,d] f32  ->  xb [B,S,d] f32 + f16
__global__ __launch_bounds__(256) void convert_x_kernel(
    const float* __restrict__ x, float* __restrict__ x32,
    _Float16* __restrict__ x16) {
  size_t i = (size_t)blockIdx.x * 256 + threadIdx.x;  // [S,B,d] flat
  int c = (int)(i % DMODEL);
  size_t sb = i / DMODEL;
  int b = (int)(sb % BATCH);
  int s = (int)(sb / BATCH);
  size_t o = ((size_t)b * SEQ + s) * DMODEL + c;
  float v = x[i];
  x32[o] = v;
  x16[o] = (_Float16)v;
}

// xb [B,S,d] f32 -> out [S,B,d] f32
__global__ __launch_bounds__(256) void out_kernel(
    const float* __restrict__ x32, float* __restrict__ out) {
  size_t i = (size_t)blockIdx.x * 256 + threadIdx.x;  // [S,B,d] flat
  int c = (int)(i % DMODEL);
  size_t sb = i / DMODEL;
  int b = (int)(sb % BATCH);
  int s = (int)(sb / BATCH);
  out[i] = x32[((size_t)b * SEQ + s) * DMODEL + c];
}

// ---------------------------------------------------------------------------
// GEMM: out[M,N] = A_f16[M,K] * W_f16[N,K]^T + bias[N]   (M == MROWS)
// 128x128 block tile, K-step 64, 8 waves of 32x64 (16 WMMAs/wave/step).
// Double-buffered LDS, tiles staged by the Tensor Data Mover when available.
// ---------------------------------------------------------------------------
template <bool OUT_F16, bool RELU>
__global__ __launch_bounds__(256) void gemm_kernel(
    const _Float16* __restrict__ A, const _Float16* __restrict__ W,
    const float* __restrict__ bias, void* __restrict__ out, int N, int K) {
  __shared__ _Float16 As[2][128][64];
  __shared__ _Float16 Bs[2][128][64];

  const int tid   = threadIdx.x;
  const int lane  = tid & 31;
  const int wave  = tid >> 5;
  const int lrow  = lane & 15;
  const int lhalf = lane >> 4;

  const int nTiles = N >> 7;
  const int mBlk = (blockIdx.x / nTiles) << 7;
  const int nBlk = (blockIdx.x % nTiles) << 7;
  const int m0 = (wave & 3) << 5;    // 4 waves along M (32 each)
  const int n0 = (wave >> 2) << 6;   // 2 waves along N (64 each)

  v8f acc[2][4];
#pragma unroll
  for (int mi = 0; mi < 2; ++mi)
#pragma unroll
    for (int ni = 0; ni < 4; ++ni)
#pragma unroll
      for (int e = 0; e < 8; ++e) acc[mi][ni][e] = 0.0f;

  const int nK = K >> 6;

#ifdef USE_TDM
  const unsigned int ldsA0 = (unsigned int)(unsigned long long)&As[0][0][0];
  const unsigned int ldsA1 = (unsigned int)(unsigned long long)&As[1][0][0];
  const unsigned int ldsB0 = (unsigned int)(unsigned long long)&Bs[0][0][0];
  const unsigned int ldsB1 = (unsigned int)(unsigned long long)&Bs[1][0][0];
  if (tid < 32) {  // wave 0 drives the TDM pipeline
    tdm_load_tile_f16(A + (size_t)mBlk * K, ldsA0, (unsigned)K, (unsigned)MROWS);
    tdm_load_tile_f16(W + (size_t)nBlk * K, ldsB0, (unsigned)K, (unsigned)N);
  }
#endif

  for (int kt = 0; kt < nK; ++kt) {
    const int cur = kt & 1;
#ifdef USE_TDM
    if (tid < 32) {
      if (kt + 1 < nK) {
        const int k1 = (kt + 1) << 6;
        tdm_load_tile_f16(A + (size_t)mBlk * K + k1, cur ? ldsA0 : ldsA1,
                          (unsigned)K, (unsigned)MROWS);
        tdm_load_tile_f16(W + (size_t)nBlk * K + k1, cur ? ldsB0 : ldsB1,
                          (unsigned)K, (unsigned)N);
        wait_tensorcnt<2>();   // next tile's 2 TDM ops may remain in flight
      } else {
        wait_tensorcnt<0>();
      }
    }
#else
    {
      const int k0 = kt << 6;
#pragma unroll
      for (int i = 0; i < 4; ++i) {
        int idx = tid + i * 256;     // 1024 v8h chunks per tile
        int row = idx >> 3;
        int ck  = (idx & 7) << 3;
        *(v8h*)(&As[cur][row][ck]) =
            *(const v8h*)(&A[(size_t)(mBlk + row) * K + k0 + ck]);
        *(v8h*)(&Bs[cur][row][ck]) =
            *(const v8h*)(&W[(size_t)(nBlk + row) * K + k0 + ck]);
      }
    }
#endif
    __syncthreads();

#pragma unroll
    for (int ks = 0; ks < 2; ++ks) {
      v16h a[2];
#pragma unroll
      for (int mi = 0; mi < 2; ++mi) {
        v8h* ap = (v8h*)&a[mi];
        const _Float16* p = &As[cur][m0 + mi * 16 + lrow][ks * 32 + lhalf * 8];
        ap[0] = *(const v8h*)(p);        // K = lhalf*8 + 0..7
        ap[1] = *(const v8h*)(p + 16);   // K = lhalf*8 + 16..23
      }
#pragma unroll
      for (int ni = 0; ni < 4; ++ni) {
        v16h b;
        v8h* bp = (v8h*)&b;
        const _Float16* p = &Bs[cur][n0 + ni * 16 + lrow][ks * 32 + lhalf * 8];
        bp[0] = *(const v8h*)(p);        // K = lhalf*8 + 0..7
        bp[1] = *(const v8h*)(p + 16);   // K = lhalf*8 + 16..23
        acc[0][ni] = wmma_f16(a[0], b, acc[0][ni]);
        acc[1][ni] = wmma_f16(a[1], b, acc[1][ni]);
      }
    }
    __syncthreads();
  }

  // epilogue: C layout M = r + 8*(lane>=16), N = lane%16
#pragma unroll
  for (int ni = 0; ni < 4; ++ni) {
    const int col = nBlk + n0 + ni * 16 + lrow;
    const float bv = bias[col];
#pragma unroll
    for (int mi = 0; mi < 2; ++mi) {
      const int rbase = mBlk + m0 + mi * 16 + lhalf * 8;
#pragma unroll
      for (int r = 0; r < 8; ++r) {
        float v = acc[mi][ni][r] + bv;
        if (RELU) v = fmaxf(v, 0.0f);
        size_t o = (size_t)(rbase + r) * N + col;
        if (OUT_F16)
          ((_Float16*)out)[o] = (_Float16)v;
        else
          ((float*)out)[o] = v;
      }
    }
  }
}

// ---------------------------------------------------------------------------
// Flash attention, mask: allowed <=> j <= i + WIN (and j < S).
// One wave per (b, head, 16-query tile). Key blocks of 32.
// ---------------------------------------------------------------------------
__global__ __launch_bounds__(128) void attn_kernel(
    const _Float16* __restrict__ qkv, _Float16* __restrict__ attn_out) {
  __shared__ _Float16 Pb[4][16][32];   // probabilities, per wave
  __shared__ _Float16 Vt[4][64][32];   // V transposed [dim][key], per wave

  const int tid   = threadIdx.x;
  const int lane  = tid & 31;
  const int wave  = tid >> 5;
  const int lrow  = lane & 15;
  const int lhalf = lane >> 4;

  const int wg   = blockIdx.x * 4 + wave;
  const int qt   = wg & 63;        // 64 query tiles of 16
  const int bh   = wg >> 6;
  const int head = bh & (NHEAD - 1);
  const int b    = bh >> 4;
  const int q0   = qt << 4;

  const _Float16* base = qkv + (size_t)b * SEQ * (3 * DMODEL);

  // Q A-fragments: row = lane%16, dims split per A layout
  v16h qa[2];
#pragma unroll
  for (int ks = 0; ks < 2; ++ks) {
    const _Float16* p =
        base + (size_t)(q0 + lrow) * (3 * DMODEL) + head * HD + ks * 32 + lhalf * 8;
    v8h* ap = (v8h*)&qa[ks];
    ap[0] = *(const v8h*)(p);
    ap[1] = *(const v8h*)(p + 16);
  }

  v8f o[4];
  float m[8], l[8];
#pragma unroll
  for (int f = 0; f < 4; ++f)
#pragma unroll
    for (int e = 0; e < 8; ++e) o[f][e] = 0.0f;
#pragma unroll
  for (int r = 0; r < 8; ++r) { m[r] = NEG_BIG; l[r] = 0.0f; }

  const int kend = (q0 + 16 + WIN) < SEQ ? (q0 + 16 + WIN) : SEQ;
  for (int k0 = 0; k0 < kend; k0 += 32) {
    // ---- scores S = Q K^T, two 16-key n-tiles, K=64 via 2 chained steps
    v8f s[2];
#pragma unroll
    for (int nt = 0; nt < 2; ++nt) {
#pragma unroll
      for (int e = 0; e < 8; ++e) s[nt][e] = 0.0f;
      int key = k0 + nt * 16 + lrow;
      int keyc = key < SEQ ? key : SEQ - 1;
      const _Float16* kp = base + (size_t)keyc * (3 * DMODEL) + DMODEL + head * HD;
#pragma unroll
      for (int ks = 0; ks < 2; ++ks) {
        v16h kb;
        v8h* bp = (v8h*)&kb;
        const _Float16* p = kp + ks * 32 + lhalf * 8;
        bp[0] = *(const v8h*)(p);
        bp[1] = *(const v8h*)(p + 16);
        s[nt] = wmma_f16(qa[ks], kb, s[nt]);
      }
    }

    // ---- stage V transposed: each lane handles a key pair x 32 dims,
    //      writing packed f16 pairs (b32 DS stores)
    {
      const int kp2 = lane & 15;
      const int d0  = (lane >> 4) * 32;
      int kA = k0 + 2 * kp2;
      int kB = kA + 1;
      kA = kA < SEQ ? kA : SEQ - 1;
      kB = kB < SEQ ? kB : SEQ - 1;
      const _Float16* va = base + (size_t)kA * (3 * DMODEL) + 2 * DMODEL + head * HD + d0;
      const _Float16* vb = base + (size_t)kB * (3 * DMODEL) + 2 * DMODEL + head * HD + d0;
#pragma unroll
      for (int c = 0; c < 4; ++c) {
        v8h ta = *(const v8h*)(va + c * 8);
        v8h tb = *(const v8h*)(vb + c * 8);
#pragma unroll
        for (int e = 0; e < 8; ++e) {
          union { _Float16 h[2]; unsigned int u; } pk;
          pk.h[0] = ta[e];
          pk.h[1] = tb[e];
          *(unsigned int*)(&Vt[wave][d0 + c * 8 + e][2 * kp2]) = pk.u;
        }
      }
    }

    // ---- online softmax (rows striped: lanes 0-15 hold M=r, 16-31 M=r+8)
#pragma unroll
    for (int r = 0; r < 8; ++r) {
      const int i  = q0 + r + lhalf * 8;
      const int j0 = k0 + lrow;
      const int j1 = j0 + 16;
      float s0 = s[0][r] * 0.125f;  // 1/sqrt(64)
      float s1 = s[1][r] * 0.125f;
      if (j0 > i + WIN || j0 >= SEQ) s0 = NEG_BIG;
      if (j1 > i + WIN || j1 >= SEQ) s1 = NEG_BIG;
      float mx = fmaxf(s0, s1);
      mx = fmaxf(mx, __shfl_xor(mx, 1, 16));
      mx = fmaxf(mx, __shfl_xor(mx, 2, 16));
      mx = fmaxf(mx, __shfl_xor(mx, 4, 16));
      mx = fmaxf(mx, __shfl_xor(mx, 8, 16));
      const float mnew  = fmaxf(m[r], mx);
      const float alpha = __expf(m[r] - mnew);
      const float p0 = __expf(s0 - mnew);
      const float p1 = __expf(s1 - mnew);
      float ps = p0 + p1;
      ps += __shfl_xor(ps, 1, 16);
      ps += __shfl_xor(ps, 2, 16);
      ps += __shfl_xor(ps, 4, 16);
      ps += __shfl_xor(ps, 8, 16);
      l[r] = l[r] * alpha + ps;
      m[r] = mnew;
#pragma unroll
      for (int f = 0; f < 4; ++f) o[f][r] *= alpha;
      Pb[wave][r + lhalf * 8][lrow]      = (_Float16)p0;
      Pb[wave][r + lhalf * 8][lrow + 16] = (_Float16)p1;
    }

    // ---- O += P (16x32) x V (32x64): re-layout P via LDS into A-fragment
    v16h pa;
    {
      v8h* ap = (v8h*)&pa;
      const _Float16* p = &Pb[wave][lrow][lhalf * 8];
      ap[0] = *(const v8h*)(p);
      ap[1] = *(const v8h*)(p + 16);
    }
#pragma unroll
    for (int f = 0; f < 4; ++f) {
      v16h vb;
      v8h* bp = (v8h*)&vb;
      const _Float16* p = &Vt[wave][f * 16 + lrow][lhalf * 8];
      bp[0] = *(const v8h*)(p);
      bp[1] = *(const v8h*)(p + 16);
      o[f] = wmma_f16(pa, vb, o[f]);
    }
  }

  // ---- normalize and write f16 output [B,S,d]
#pragma unroll
  for (int f = 0; f < 4; ++f) {
    const int col = head * HD + f * 16 + lrow;
#pragma unroll
    for (int r = 0; r < 8; ++r) {
      const int row = q0 + r + lhalf * 8;
      attn_out[(size_t)(b * SEQ + row) * DMODEL + col] = (_Float16)(o[f][r] / l[r]);
    }
  }
}

// ---------------------------------------------------------------------------
// y = LayerNorm(gin + res) * gamma + beta ; write f32 residual + f16 input
// one block per row of 1024
// ---------------------------------------------------------------------------
__global__ __launch_bounds__(256) void add_ln_kernel(
    const float* __restrict__ gin, const float* __restrict__ res,
    const float* __restrict__ gamma, const float* __restrict__ beta,
    float* __restrict__ x32, _Float16* __restrict__ x16) {
  __shared__ float s1[256];
  __shared__ float s2[256];
  const int row = blockIdx.x;
  const int tid = threadIdx.x;
  float v[4];
  float sum = 0.0f, sq = 0.0f;
#pragma unroll
  for (int i = 0; i < 4; ++i) {
    const int c = tid + i * 256;
    v[i] = gin[(size_t)row * DMODEL + c] + res[(size_t)row * DMODEL + c];
    sum += v[i];
    sq  += v[i] * v[i];
  }
  s1[tid] = sum; s2[tid] = sq;
  __syncthreads();
  for (int off = 128; off > 0; off >>= 1) {
    if (tid < off) { s1[tid] += s1[tid + off]; s2[tid] += s2[tid + off]; }
    __syncthreads();
  }
  const float mu  = s1[0] * (1.0f / DMODEL);
  const float var = s2[0] * (1.0f / DMODEL) - mu * mu;
  const float rs  = rsqrtf(var + 1e-5f);
#pragma unroll
  for (int i = 0; i < 4; ++i) {
    const int c = tid + i * 256;
    const float y = (v[i] - mu) * rs * gamma[c] + beta[c];
    x32[(size_t)row * DMODEL + c] = y;
    x16[(size_t)row * DMODEL + c] = (_Float16)y;
  }
}

// ---------------------------------------------------------------------------
extern "C" void kernel_launch(void* const* d_in, const int* in_sizes, int n_in,
                              void* d_out, int out_size, void* d_ws,
                              size_t ws_size, hipStream_t stream) {
  const float* x    = (const float*)d_in[0];
  const float* Wqkv = (const float*)d_in[1];
  const float* bqkv = (const float*)d_in[2];
  const float* Wo   = (const float*)d_in[3];
  const float* bo   = (const float*)d_in[4];
  const float* W1   = (const float*)d_in[5];
  const float* b1   = (const float*)d_in[6];
  const float* W2   = (const float*)d_in[7];
  const float* b2   = (const float*)d_in[8];
  const float* g1   = (const float*)d_in[9];
  const float* be1  = (const float*)d_in[10];
  const float* g2   = (const float*)d_in[11];
  const float* be2  = (const float*)d_in[12];

  char* ws = (char*)d_ws;
  const size_t MB = 1u << 20;
  float*    xb32   = (float*)(ws + 0);             // 16 MB
  _Float16* xb16   = (_Float16*)(ws + 16 * MB);    //  8 MB
  _Float16* qkv    = (_Float16*)(ws + 24 * MB);    // 24 MB
  _Float16* attn   = (_Float16*)(ws + 48 * MB);    //  8 MB
  _Float16* ff     = (_Float16*)(ws + 56 * MB);    // 32 MB
  float*    tmp    = (float*)(ws + 88 * MB);       // 16 MB
  _Float16* wqkv16 = (_Float16*)(ws + 104 * MB);   //  6 MB
  _Float16* wo16   = (_Float16*)(ws + 110 * MB);   //  2 MB
  _Float16* w116   = (_Float16*)(ws + 112 * MB);   //  8 MB
  _Float16* w216   = (_Float16*)(ws + 120 * MB);   //  8 MB (total 128 MB)

  const int elemBlocks = (MROWS * DMODEL) / 256;  // 16384
  convert_x_kernel<<<elemBlocks, 256, 0, stream>>>(x, xb32, xb16);

  for (int l = 0; l < 4; ++l) {
    // per-layer weight conversion to f16 (keeps [N,K] layout for TDM tiles)
    f32_to_f16_kernel<<<(3 * DMODEL * DMODEL) / 256, 256, 0, stream>>>(
        Wqkv + (size_t)l * 3 * DMODEL * DMODEL, wqkv16);
    f32_to_f16_kernel<<<(DMODEL * DMODEL) / 256, 256, 0, stream>>>(
        Wo + (size_t)l * DMODEL * DMODEL, wo16);
    f32_to_f16_kernel<<<(DFF * DMODEL) / 256, 256, 0, stream>>>(
        W1 + (size_t)l * DFF * DMODEL, w116);
    f32_to_f16_kernel<<<(DMODEL * DFF) / 256, 256, 0, stream>>>(
        W2 + (size_t)l * DMODEL * DFF, w216);

    // QKV projection -> qkv f16 [B*S, 3d]
    gemm_kernel<true, false>
        <<<(MROWS / 128) * (3 * DMODEL / 128), 256, 0, stream>>>(
            xb16, wqkv16, bqkv + l * 3 * DMODEL, qkv, 3 * DMODEL, DMODEL);

    // banded-causal flash attention -> attn f16 [B*S, d]
    attn_kernel<<<(BATCH * NHEAD * (SEQ / 16)) / 4, 128, 0, stream>>>(qkv, attn);

    // output projection -> tmp f32
    gemm_kernel<false, false>
        <<<(MROWS / 128) * (DMODEL / 128), 256, 0, stream>>>(
            attn, wo16, bo + l * DMODEL, tmp, DMODEL, DMODEL);
    add_ln_kernel<<<MROWS, 256, 0, stream>>>(tmp, xb32, g1 + l * DMODEL,
                                             be1 + l * DMODEL, xb32, xb16);

    // FFN
    gemm_kernel<true, true>
        <<<(MROWS / 128) * (DFF / 128), 256, 0, stream>>>(
            xb16, w116, b1 + l * DFF, ff, DFF, DMODEL);
    gemm_kernel<false, false>
        <<<(MROWS / 128) * (DMODEL / 128), 256, 0, stream>>>(
            ff, w216, b2 + l * DMODEL, tmp, DMODEL, DFF);
    add_ln_kernel<<<MROWS, 256, 0, stream>>>(tmp, xb32, g2 + l * DMODEL,
                                             be2 + l * DMODEL, xb32, xb16);
  }

  out_kernel<<<elemBlocks, 256, 0, stream>>>(xb32, (float*)d_out);
}
